// ContentAdaptiveDegradation_50474455662797
// MI455X (gfx1250) — compile-verified, matrix operand order checked by reading, not run on hardware
//
#include <hip/hip_runtime.h>
#include <hip/hip_bf16.h>
#include <math.h>

typedef __attribute__((ext_vector_type(16))) _Float16 v16h;
typedef __attribute__((ext_vector_type(8)))  float    v8f;
typedef __attribute__((ext_vector_type(4)))  unsigned int v4u;
typedef __attribute__((ext_vector_type(4)))  int      v4i;
typedef __attribute__((ext_vector_type(8)))  int      v8i;

#define WAVES_PER_BLOCK 8

// ---------------------------------------------------------------------------
// Implicit-GEMM conv 3x3, stride 2, pad 1 via WMMA f32_16x16x32_f16 (wave32).
// A = weights [COUT x (CIN*9)], B = im2col(input). Each wave computes a
// 16(M) x 64(N) strip: the A fragment is built once per 32-K step and reused
// by 4 WMMAs (4 N-tiles), amortizing weight loads and K->(ci,ky,kx) decode.
// Optionally fuses BN(scale,shift)+ReLU of the PREVIOUS layer into the input
// load. Writes raw conv output (+bias) in fp32 for batch-stats computation.
// ---------------------------------------------------------------------------
template <int CIN, int COUT, int OH, bool FUSE_BN_RELU>
__global__ __launch_bounds__(256) void conv_wmma_kernel(
    const float* __restrict__ in,     // [B, CIN, 2*OH, 2*OH]
    const float* __restrict__ w,      // [COUT, CIN, 3, 3]
    const float* __restrict__ bias,   // [COUT]
    const float* __restrict__ scale,  // [CIN] fused BN scale for input
    const float* __restrict__ shift,  // [CIN]
    float* __restrict__ out)          // [B, COUT, OH, OH]
{
    constexpr int IH  = OH * 2;
    constexpr int KT  = CIN * 9;
    constexpr int KTP = ((KT + 31) / 32) * 32;
    constexpr int NT  = (OH * OH) / 16;  // 16-wide N tiles per image
    constexpr int NB  = 4;               // N tiles per wave
    constexpr int NTB = NT / NB;
    constexpr int MT  = COUT / 16;

    const int wave = blockIdx.x * WAVES_PER_BLOCK + (threadIdx.x >> 5);
    const int lane = threadIdx.x & 31;
    const int ntb = wave % NTB;
    const int mt  = (wave / NTB) % MT;
    const int b   = wave / (NTB * MT);

    const int ln   = lane & 15;   // N (and A's M) sub-index
    const int kgrp = lane >> 4;   // which K half-group this lane holds

    int oy[NB], ox[NB];
#pragma unroll
    for (int j = 0; j < NB; ++j) {
        const int n = (ntb * NB + j) * 16 + ln;
        oy[j] = n / OH;
        ox[j] = n % OH;
    }

    v8f acc[NB] = {};
    for (int kk = 0; kk < KTP; kk += 32) {
        // ---- A fragment (shared across the 4 N tiles)
        v16h afrag;
#pragma unroll
        for (int e = 0; e < 16; ++e) {
            const int K = kk + ((e < 8) ? 0 : 16) + kgrp * 8 + (e & 7);
            float av = 0.0f;
            if ((KT % 32 == 0) || (K < KT)) av = w[(mt * 16 + ln) * KT + K];
            afrag[e] = (_Float16)av;
        }
#pragma unroll
        for (int j = 0; j < NB; ++j) {
            v16h bfrag;
#pragma unroll
            for (int e = 0; e < 16; ++e) {
                const int K = kk + ((e < 8) ? 0 : 16) + kgrp * 8 + (e & 7);
                float bv = 0.0f;
                if ((KT % 32 == 0) || (K < KT)) {
                    const int ci = K / 9;
                    const int r9 = K % 9;
                    const int ky = r9 / 3;
                    const int kx = r9 % 3;
                    const int iy = oy[j] * 2 - 1 + ky;
                    const int ix = ox[j] * 2 - 1 + kx;
                    if (iy >= 0 && iy < IH && ix >= 0 && ix < IH) {
                        float raw = in[((size_t)(b * CIN + ci) * IH + iy) * IH + ix];
                        if (FUSE_BN_RELU)
                            raw = fmaxf(fmaf(raw, scale[ci], shift[ci]), 0.0f);
                        bv = raw;
                    }
                }
                bfrag[e] = (_Float16)bv;
            }
            acc[j] = __builtin_amdgcn_wmma_f32_16x16x32_f16(
                false, afrag, false, bfrag, (short)0, acc[j], false, false);
        }
    }

    // C/D layout: VGPR r -> M = r + 8*kgrp, N = lane&15
#pragma unroll
    for (int j = 0; j < NB; ++j) {
        const int n = (ntb * NB + j) * 16 + ln;
#pragma unroll
        for (int r = 0; r < 8; ++r) {
            const int m = mt * 16 + r + 8 * kgrp;
            out[(size_t)(b * COUT + m) * (OH * OH) + n] = acc[j][r] + bias[m];
        }
    }
}

// ---------------------------------------------------------------------------
// Deterministic per-channel batch statistics: stage 1 (block partials).
// ---------------------------------------------------------------------------
template <int C, int HW, int NB, int CHUNKS>
__global__ __launch_bounds__(256) void stats_kernel(
    const float* __restrict__ data, float* __restrict__ partials)
{
    const int c = blockIdx.x;
    const int chunk = blockIdx.y;
    constexpr int CHW = HW / CHUNKS;
    const int tid = threadIdx.x;

    float s = 0.0f, ss = 0.0f;
    for (int b = 0; b < NB; ++b) {
        const float* p = data + ((size_t)b * C + c) * HW + (size_t)chunk * CHW;
        for (int i = tid; i < CHW; i += 256) {
            float v = p[i];
            s += v;
            ss += v * v;
        }
    }
    __shared__ float sh[256], sh2[256];
    sh[tid] = s; sh2[tid] = ss;
    __syncthreads();
    for (int o = 128; o > 0; o >>= 1) {
        if (tid < o) { sh[tid] += sh[tid + o]; sh2[tid] += sh2[tid + o]; }
        __syncthreads();
    }
    if (tid == 0) {
        partials[(c * CHUNKS + chunk) * 2 + 0] = sh[0];
        partials[(c * CHUNKS + chunk) * 2 + 1] = sh2[0];
    }
}

// Stage 2: fold partials in fixed order -> fused BN scale/shift per channel.
template <int C, int CHUNKS>
__global__ void finalize_stats_kernel(
    const float* __restrict__ partials,
    const float* __restrict__ gamma, const float* __restrict__ beta,
    float* __restrict__ scale, float* __restrict__ shift, float count)
{
    const int c = threadIdx.x;
    if (c >= C) return;
    float s = 0.0f, ss = 0.0f;
    for (int i = 0; i < CHUNKS; ++i) {
        s  += partials[(c * CHUNKS + i) * 2 + 0];
        ss += partials[(c * CHUNKS + i) * 2 + 1];
    }
    const float mu  = s / count;
    const float var = ss / count - mu * mu;
    const float inv = rsqrtf(var + 1e-5f);
    const float sc  = gamma[c] * inv;
    scale[c] = sc;
    shift[c] = beta[c] - mu * sc;
}

// ---------------------------------------------------------------------------
// Global average pool of relu(bn(conv3raw)) -> feat [B,128]. grid=(128,B).
// ---------------------------------------------------------------------------
__global__ __launch_bounds__(256) void gap_kernel(
    const float* __restrict__ conv3raw,
    const float* __restrict__ scale, const float* __restrict__ shift,
    float* __restrict__ feat)
{
    const int c = blockIdx.x;
    const int b = blockIdx.y;
    const int tid = threadIdx.x;
    const float* p = conv3raw + ((size_t)b * 128 + c) * 4096;
    const float sc = scale[c], sf = shift[c];
    float s = 0.0f;
    for (int i = tid; i < 4096; i += 256)
        s += fmaxf(fmaf(p[i], sc, sf), 0.0f);
    __shared__ float sh[256];
    sh[tid] = s;
    __syncthreads();
    for (int o = 128; o > 0; o >>= 1) {
        if (tid < o) sh[tid] += sh[tid + o];
        __syncthreads();
    }
    if (tid == 0) feat[b * 128 + c] = sh[0] * (1.0f / 4096.0f);
}

// ---------------------------------------------------------------------------
// fc + gumbel softmax (tau=1, soft). One block. weights -> d_out + ws.
// ---------------------------------------------------------------------------
__global__ void fc_gumbel_kernel(
    const float* __restrict__ feat, const float* __restrict__ fc_w,
    const float* __restrict__ fc_b, const float* __restrict__ gumbel,
    float* __restrict__ out_w, float* __restrict__ ws_w)
{
    const int tid = threadIdx.x;
    __shared__ float lg[16][10];
    __shared__ float mx[16], sm[16];
    if (tid < 160) {
        const int b = tid / 10, k = tid % 10;
        float s = fc_b[k] + gumbel[b * 10 + k];
        for (int c = 0; c < 128; ++c)
            s = fmaf(feat[b * 128 + c], fc_w[k * 128 + c], s);
        lg[b][k] = s;
    }
    __syncthreads();
    if (tid < 16) {
        float m = -INFINITY;
        for (int k = 0; k < 10; ++k) m = fmaxf(m, lg[tid][k]);
        float s = 0.0f;
        for (int k = 0; k < 10; ++k) s += expf(lg[tid][k] - m);
        mx[tid] = m; sm[tid] = s;
    }
    __syncthreads();
    if (tid < 160) {
        const int b = tid / 10, k = tid % 10;
        const float w = expf(lg[b][k] - mx[b]) / sm[b];
        out_w[tid] = w;
        ws_w[tid] = w;
    }
}

// ---------------------------------------------------------------------------
// Softmax over 13x13 per (k,c) row of the kernel bank. grid = 50 rows.
// ---------------------------------------------------------------------------
__global__ void knorm_kernel(const float* __restrict__ kb,
                             float* __restrict__ out_kn,
                             float* __restrict__ ws_kn)
{
    const int row = blockIdx.x;       // 0..49 = k*5+c
    const int tid = threadIdx.x;      // 256
    __shared__ float red[256];
    float v = (tid < 169) ? kb[row * 169 + tid] : -INFINITY;
    red[tid] = v;
    __syncthreads();
    for (int o = 128; o > 0; o >>= 1) {
        if (tid < o) red[tid] = fmaxf(red[tid], red[tid + o]);
        __syncthreads();
    }
    const float m = red[0];
    __syncthreads();
    float e = (tid < 169) ? expf(v - m) : 0.0f;
    red[tid] = e;
    __syncthreads();
    for (int o = 128; o > 0; o >>= 1) {
        if (tid < o) red[tid] += red[tid + o];
        __syncthreads();
    }
    const float s = red[0];
    if (tid < 169) {
        const float o = e / s;
        out_kn[row * 169 + tid] = o;
        ws_kn[row * 169 + tid] = o;
    }
}

// ---------------------------------------------------------------------------
// Per-sample kernel / sigma synthesis. One block.
// ---------------------------------------------------------------------------
__global__ void synth_kernel(const float* __restrict__ wts,   // [16,10]
                             const float* __restrict__ knorm, // [10,5,169]
                             const float* __restrict__ sigma_bank, // [10,5]
                             float* __restrict__ bk,  // [16,5,169]
                             float* __restrict__ bs)  // [16,5]
{
    const int tid = threadIdx.x;
    for (int idx = tid; idx < 16 * 5 * 169; idx += 256) {
        const int b = idx / 845;
        const int rem = idx % 845;
        const int c = rem / 169;
        const int p = rem % 169;
        float s = 0.0f;
        for (int k = 0; k < 10; ++k)
            s = fmaf(wts[b * 10 + k], knorm[(k * 5 + c) * 169 + p], s);
        bk[idx] = s;
    }
    for (int idx = tid; idx < 80; idx += 256) {
        const int b = idx / 5, c = idx % 5;
        float s = 0.0f;
        for (int k = 0; k < 10; ++k)
            s = fmaf(wts[b * 10 + k], log1pf(expf(sigma_bank[k * 5 + c])), s);
        bs[idx] = s;
    }
}

// ---------------------------------------------------------------------------
// Depthwise 13x13 conv evaluated ONLY at the ::4 strided outputs (128x128).
// Interior tiles: 73x73 fp32 input tile DMA'd to LDS by the Tensor Data
// Mover (2D D#, tensor dims clipped to the remaining extent so right/bottom
// overhang reads return zero per ISA 8.2). Top/left halo tiles (negative
// origin) use the manual bounds-checked load path. grid = (B*C, 64 tiles).
// ---------------------------------------------------------------------------
__global__ __launch_bounds__(256) void depthwise_kernel(
    const float* __restrict__ x,      // [B,C,512,512]
    const float* __restrict__ bk,     // [B,C,169]
    const float* __restrict__ bs,     // [B,C]
    const float* __restrict__ noise,  // [B,C,128,128]
    float* __restrict__ out)          // [B,C,128,128]
{
    const int bc   = blockIdx.x;      // 0..79
    const int tile = blockIdx.y;      // 0..63
    const int ty0 = (tile / 8) * 16;
    const int tx0 = (tile % 8) * 16;
    const int tid = threadIdx.x;

    __shared__ float kker[169];
    __shared__ float tilebuf[73 * 73];
    if (tid < 169) kker[tid] = bk[bc * 169 + tid];

    const float* xb = x + (size_t)bc * 512 * 512;
    const int iy0 = 4 * ty0 - 6;
    const int ix0 = 4 * tx0 - 6;

    if (iy0 >= 0 && ix0 >= 0) {
        // ---- TDM path: one wave issues the 2D tile DMA, waits TENSORcnt.
        if (tid < 32) {
            const unsigned long long ga =
                (unsigned long long)(size_t)(xb + (size_t)iy0 * 512 + ix0);
            const unsigned int lds_addr =
                (unsigned int)(unsigned long long)(size_t)(void*)tilebuf;
            const unsigned int tdim0 = (unsigned int)(512 - ix0); // remaining X
            const unsigned int tdim1 = (unsigned int)(512 - iy0); // remaining Y
            v4u g0;
            g0[0] = 1u;                                         // count=1
            g0[1] = lds_addr;                                   // lds_addr
            g0[2] = (unsigned int)(ga & 0xFFFFFFFFull);         // gaddr[31:0]
            g0[3] = (unsigned int)((ga >> 32) & 0x1FFFFFFull)   // gaddr[56:32]
                    | (2u << 30);                               // type=2 (image)
            v8i g1;
            g1[0] = (int)(2u << 16);                            // data_size=4B
            g1[1] = (int)((tdim0 & 0xFFFFu) << 16);             // tensor_dim0 lo
            g1[2] = (int)(((tdim0 >> 16) & 0xFFFFu) |
                          ((tdim1 & 0xFFFFu) << 16));           // dim0 hi|dim1 lo
            g1[3] = (int)(((tdim1 >> 16) & 0xFFFFu) |
                          (73u << 16));                         // dim1 hi|tile0=73
            g1[4] = (int)73u;                                   // tile1=73, tile2=0
            g1[5] = (int)512u;                                  // dim0_stride lo
            g1[6] = 0;                                          // stride hi | dim1_stride
            g1[7] = 0;
            v4i gz4 = {};
            v8i gz8 = {};
            __builtin_amdgcn_tensor_load_to_lds(g0, g1, gz4, gz4, gz8, 0);
            __builtin_amdgcn_s_wait_tensorcnt(0);
        }
    } else {
        // ---- Halo tiles: manual bounds-checked cooperative load.
        for (int i = tid; i < 73 * 73; i += 256) {
            const int r = i / 73, cc = i % 73;
            const int iy = iy0 + r, ix = ix0 + cc;
            float v = 0.0f;
            if (iy >= 0 && iy < 512 && ix >= 0 && ix < 512) v = xb[iy * 512 + ix];
            tilebuf[i] = v;
        }
    }
    __syncthreads();

    const int ly = 4 * (tid / 16);
    const int lx = 4 * (tid % 16);
    float acc = 0.0f;
#pragma unroll
    for (int ky = 0; ky < 13; ++ky)
#pragma unroll
        for (int kx = 0; kx < 13; ++kx)
            acc = fmaf(tilebuf[(ly + ky) * 73 + lx + kx], kker[ky * 13 + kx], acc);

    const int oy = ty0 + tid / 16;
    const int ox = tx0 + tid % 16;
    const size_t o = (size_t)bc * (128 * 128) + oy * 128 + ox;
    out[o] = acc + noise[o] * bs[bc];
}

// ---------------------------------------------------------------------------
extern "C" void kernel_launch(void* const* d_in, const int* in_sizes, int n_in,
                              void* d_out, int out_size, void* d_ws, size_t ws_size,
                              hipStream_t stream)
{
    const float* x          = (const float*)d_in[0];
    const float* conv1_w    = (const float*)d_in[1];
    const float* conv1_b    = (const float*)d_in[2];
    const float* bn1_g      = (const float*)d_in[3];
    const float* bn1_b      = (const float*)d_in[4];
    const float* conv2_w    = (const float*)d_in[5];
    const float* conv2_b    = (const float*)d_in[6];
    const float* bn2_g      = (const float*)d_in[7];
    const float* bn2_b      = (const float*)d_in[8];
    const float* conv3_w    = (const float*)d_in[9];
    const float* conv3_b    = (const float*)d_in[10];
    const float* bn3_g      = (const float*)d_in[11];
    const float* bn3_b      = (const float*)d_in[12];
    const float* fc_w       = (const float*)d_in[13];
    const float* fc_b       = (const float*)d_in[14];
    const float* kernel_bank= (const float*)d_in[15];
    const float* sigma_bank = (const float*)d_in[16];
    const float* gumbel_g   = (const float*)d_in[17];
    const float* noise      = (const float*)d_in[18];

    float* ws = (float*)d_ws;
    // workspace layout (floats)
    const size_t o_c1   = 0;                         // 16*32*256*256
    const size_t o_c2   = o_c1 + 33554432;           // 16*64*128*128
    const size_t o_c3   = o_c2 + 16777216;           // 16*128*64*64
    const size_t o_part = o_c3 + 8388608;            // 128*32*2 partials (reused)
    const size_t o_sc1  = o_part + 8192;
    const size_t o_sf1  = o_sc1 + 32;
    const size_t o_sc2  = o_sf1 + 32;
    const size_t o_sf2  = o_sc2 + 64;
    const size_t o_sc3  = o_sf2 + 64;
    const size_t o_sf3  = o_sc3 + 128;
    const size_t o_feat = o_sf3 + 128;               // feat[16*128]
    const size_t o_w    = o_feat + 2048;             // weights[160]
    const size_t o_kn   = o_w + 160;                 // k_norm[8450]
    const size_t o_bk   = o_kn + 8450;               // batch_kernels[13520]
    const size_t o_bs   = o_bk + 13520;              // batch_sigmas[80]
    (void)o_bs; (void)ws_size; (void)n_in; (void)in_sizes; (void)out_size;

    float* out_main = (float*)d_out;                 // [16,5,128,128]
    float* out_w    = out_main + 1310720;            // [16,10]
    float* out_kn   = out_w + 160;                   // [10,5,13,13]

    // 1) kernel-bank softmax (independent of everything else)
    knorm_kernel<<<50, 256, 0, stream>>>(kernel_bank, out_kn, ws + o_kn);

    // 2) conv1 (raw input, no BN fuse): 32768 waves, 4 N-tiles/wave
    conv_wmma_kernel<5, 32, 256, false><<<4096, 256, 0, stream>>>(
        x, conv1_w, conv1_b, ws + o_sc1, ws + o_sf1, ws + o_c1);
    stats_kernel<32, 65536, 16, 32><<<dim3(32, 32), 256, 0, stream>>>(ws + o_c1, ws + o_part);
    finalize_stats_kernel<32, 32><<<1, 32, 0, stream>>>(
        ws + o_part, bn1_g, bn1_b, ws + o_sc1, ws + o_sf1, 16.0f * 65536.0f);

    // 3) conv2 with BN1+ReLU fused into input load
    conv_wmma_kernel<32, 64, 128, true><<<2048, 256, 0, stream>>>(
        ws + o_c1, conv2_w, conv2_b, ws + o_sc1, ws + o_sf1, ws + o_c2);
    stats_kernel<64, 16384, 16, 32><<<dim3(64, 32), 256, 0, stream>>>(ws + o_c2, ws + o_part);
    finalize_stats_kernel<64, 32><<<1, 64, 0, stream>>>(
        ws + o_part, bn2_g, bn2_b, ws + o_sc2, ws + o_sf2, 16.0f * 16384.0f);

    // 4) conv3 with BN2+ReLU fused
    conv_wmma_kernel<64, 128, 64, true><<<1024, 256, 0, stream>>>(
        ws + o_c2, conv3_w, conv3_b, ws + o_sc2, ws + o_sf2, ws + o_c3);
    stats_kernel<128, 4096, 16, 32><<<dim3(128, 32), 256, 0, stream>>>(ws + o_c3, ws + o_part);
    finalize_stats_kernel<128, 32><<<1, 128, 0, stream>>>(
        ws + o_part, bn3_g, bn3_b, ws + o_sc3, ws + o_sf3, 16.0f * 4096.0f);

    // 5) global average pool (BN3+ReLU fused) -> feat
    gap_kernel<<<dim3(128, 16), 256, 0, stream>>>(ws + o_c3, ws + o_sc3, ws + o_sf3, ws + o_feat);

    // 6) fc + gumbel softmax -> weights
    fc_gumbel_kernel<<<1, 256, 0, stream>>>(ws + o_feat, fc_w, fc_b, gumbel_g, out_w, ws + o_w);

    // 7) per-sample kernel / sigma synthesis
    synth_kernel<<<1, 256, 0, stream>>>(ws + o_w, ws + o_kn, sigma_bank, ws + o_bk, ws + o_bs);

    // 8) depthwise conv at strided outputs + noise (TDM tile loads)
    depthwise_kernel<<<dim3(80, 64), 256, 0, stream>>>(x, ws + o_bk, ws + o_bs, noise, out_main);
}
